// NF4Linear_11192684773386
// MI455X (gfx1250) — matrix operand order repaired and checked by reading, not compiled, not run on hardware
//
#include <hip/hip_runtime.h>
#include <hip/hip_bf16.h>
#include <stdint.h>

// ---------------------------------------------------------------------------
// NF4 Linear: out[b,s,o] = sum_i x[b,s,i] * dequant(w)[o,i]
// M = 4*2048 = 8192, K = 4096, N = 11008
// Pass 1: dequant qweight (int32 nibble indices) -> bf16 W in workspace
// Pass 2: convert x fp32 -> bf16 in workspace
// Pass 3: bf16 WMMA GEMM (v_wmma_f32_16x16x32_bf16), double-buffered
//         async global->LDS staging (ASYNCcnt pipelined)
// ---------------------------------------------------------------------------

#define OUT_DIM 11008
#define IN_DIM  4096
#define M_DIM   8192                 // 4 * 2048
#define NBLOCKS (OUT_DIM * IN_DIM / 64)

typedef __attribute__((ext_vector_type(16))) __bf16 v16bf;
typedef __attribute__((ext_vector_type(8)))  __bf16 v8bf;
typedef __attribute__((ext_vector_type(4)))  __bf16 v4bf;
typedef __attribute__((ext_vector_type(8)))  float  v8f;
typedef __attribute__((ext_vector_type(4)))  float  v4f;
typedef __attribute__((ext_vector_type(4)))  int    v4i;

__device__ const float NF4C[16] = {
    -1.0f, -0.6961928009986877f, -0.5250730514526367f, -0.39491748809814453f,
    -0.28444138169288635f, -0.18477343022823334f, -0.10644006729125977f,
    -0.029167551919817924f, 0.0f, 0.07958029955625534f, 0.16093020141124725f,
    0.24611230194568634f, 0.33791524171829224f, 0.44070982933044434f,
    0.5626170039176941f, 0.7229568362236023f};

// ------------------------- async global -> LDS helpers ----------------------
__device__ __forceinline__ void async_load_b128(unsigned lds_off, const void* gaddr) {
    asm volatile("global_load_async_to_lds_b128 %0, %1, off"
                 :: "v"(lds_off), "v"((unsigned long long)(uintptr_t)gaddr)
                 : "memory");
}
__device__ __forceinline__ void wait_asynccnt0() {
    asm volatile("s_wait_asynccnt 0x0" ::: "memory");
}
__device__ __forceinline__ void wait_asynccnt8() {
    asm volatile("s_wait_asynccnt 0x8" ::: "memory");
}

// ----------------------------- Pass 1: dequant ------------------------------
// 4 weight elements per thread (all within one 64-elem quant block).
__global__ __launch_bounds__(256) void nf4_dequant_kernel(
    const int* __restrict__ qweight,       // [OUT*IN] nibble values 0..15
    const int* __restrict__ qscalers,      // [NBLOCKS]
    const float* __restrict__ qfactor,     // [NBLOCKS/256]
    const float* __restrict__ smean,       // [1]
    __bf16* __restrict__ wout)             // [OUT*IN] bf16
{
    __shared__ float tbl[16];
    if (threadIdx.x < 16) tbl[threadIdx.x] = NF4C[threadIdx.x];
    __syncthreads();

    const long long g = (long long)blockIdx.x * blockDim.x + threadIdx.x; // quad id
    const long long blk = g >> 4;                                         // /64 elems
    const float scaler = (float)qscalers[blk] / qfactor[blk >> 8] + smean[0];

    v4i q = ((const v4i*)qweight)[g];
    v4bf o;
    o.x = (__bf16)(tbl[q.x & 15] * scaler);
    o.y = (__bf16)(tbl[q.y & 15] * scaler);
    o.z = (__bf16)(tbl[q.z & 15] * scaler);
    o.w = (__bf16)(tbl[q.w & 15] * scaler);
    ((v4bf*)wout)[g] = o;
}

// --------------------------- Pass 2: x -> bf16 ------------------------------
__global__ __launch_bounds__(256) void cvt_x_kernel(const float* __restrict__ x,
                                                    __bf16* __restrict__ xbf)
{
    const long long g = (long long)blockIdx.x * blockDim.x + threadIdx.x;
    v4f v = ((const v4f*)x)[g];
    v4bf o;
    o.x = (__bf16)v.x; o.y = (__bf16)v.y; o.z = (__bf16)v.z; o.w = (__bf16)v.w;
    ((v4bf*)xbf)[g] = o;
}

// --------------------------- Pass 3: WMMA GEMM ------------------------------
// Block tile 128(M) x 128(N), K-tile 64. 8 waves: wave grid 4(M) x 2(N),
// each wave 32x64 = 2x4 fragments of 16x16, two k=32 WMMA steps per K-tile.
// Double-buffered LDS: prefetch tile kt+1 via async-to-LDS while computing kt.
#define KTILE 64
#define LDSTR 72   // bf16 elems per LDS row: 64 + 8 pad (144 B, multiple of 16)
#define MAT_ELEMS   (128 * LDSTR)          // one A or B tile
#define PHASE_ELEMS (2 * MAT_ELEMS)        // A + B for one phase
#define MAT_BYTES   (MAT_ELEMS * 2)
#define PHASE_BYTES (PHASE_ELEMS * 2)

union ABu { v16bf v; v8bf h[2]; };

__device__ __forceinline__ void issue_tile_loads(
    const __bf16* __restrict__ A, const __bf16* __restrict__ B,
    unsigned ldsBase, int phase,
    long long rowBaseA, long long rowBaseB, int k0, int tid)
{
    const unsigned pbase = ldsBase + (unsigned)(phase * PHASE_BYTES);
#pragma unroll
    for (int i = 0; i < 4; ++i) {
        const int c   = tid + 256 * i;
        const int row = c >> 3;       // 0..127
        const int cc  = c & 7;        // 16B chunk within row
        const unsigned loff = (unsigned)(row * (LDSTR * 2) + cc * 16);
        async_load_b128(pbase + loff,
                        A + (rowBaseA + row) * IN_DIM + k0 + cc * 8);
        async_load_b128(pbase + (unsigned)MAT_BYTES + loff,
                        B + (rowBaseB + row) * IN_DIM + k0 + cc * 8);
    }
}

__global__ __launch_bounds__(256) void nf4_gemm_wmma_kernel(
    const __bf16* __restrict__ A,    // [M_DIM][IN_DIM] bf16 (x)
    const __bf16* __restrict__ B,    // [OUT_DIM][IN_DIM] bf16 (w)
    float* __restrict__ C)           // [M_DIM][OUT_DIM] fp32
{
    __shared__ __bf16 smem[2 * PHASE_ELEMS];   // 73,728 B: 2 phases x (A+B)
    const unsigned ldsBase = (unsigned)(uintptr_t)smem;

    const int tid   = threadIdx.x;
    const int lane  = tid & 31;
    const int l16   = lane & 15;
    const int lhalf = lane >> 4;          // 0: lanes 0-15, 1: lanes 16-31
    const int wv    = tid >> 5;           // 0..7
    const int waveM = wv & 3;             // 4 blocks of 32 rows
    const int waveN = wv >> 2;            // 2 blocks of 64 cols

    const int tileN = blockIdx.x;         // 0..85
    const int tileM = blockIdx.y;         // 0..63
    const long long rowBaseA = (long long)tileM * 128;
    const long long rowBaseB = (long long)tileN * 128;

    v8f acc[2][4];
#pragma unroll
    for (int mt = 0; mt < 2; ++mt)
#pragma unroll
        for (int nt = 0; nt < 4; ++nt)
            acc[mt][nt] = (v8f){0.f, 0.f, 0.f, 0.f, 0.f, 0.f, 0.f, 0.f};

    const int NT = IN_DIM / KTILE;        // 64

    // Prologue: stage tile 0 into phase 0.
    issue_tile_loads(A, B, ldsBase, 0, rowBaseA, rowBaseB, 0, tid);

    for (int kt = 0; kt < NT; ++kt) {
        const int p = kt & 1;
        if (kt + 1 < NT) {
            // Prefetch next tile into the other phase (its readers finished at
            // the end-of-iteration barrier of kt-1), then wait only for the
            // current tile's 8 loads (in-order completion => <=8 outstanding).
            issue_tile_loads(A, B, ldsBase, p ^ 1, rowBaseA, rowBaseB,
                             (kt + 1) * KTILE, tid);
            wait_asynccnt8();
        } else {
            wait_asynccnt0();
        }
        __syncthreads();

        const __bf16* As = smem + p * PHASE_ELEMS;
        const __bf16* Bs = As + MAT_ELEMS;

        // ---- two k=32 WMMA steps ----
#pragma unroll
        for (int kk = 0; kk < 2; ++kk) {
            // A fragments: lane l<16 -> M=l; VGPR0-3: K=kb..kb+7, VGPR4-7: K=kb+16..kb+23
            v16bf afr[2];
#pragma unroll
            for (int mt = 0; mt < 2; ++mt) {
                const int row = waveM * 32 + mt * 16 + l16;
                const int kb  = kk * 32 + lhalf * 8;
                ABu u;
                u.h[0] = *(const v8bf*)&As[row * LDSTR + kb];
                u.h[1] = *(const v8bf*)&As[row * LDSTR + kb + 16];
                afr[mt] = u.v;
            }
            // B fragments: col N=l16; VGPR0-3: K=kb..kb+7, VGPR4-7: K=kb+16..kb+23 (kb = 8*lhalf)
            v16bf bfr[4];
#pragma unroll
            for (int nt = 0; nt < 4; ++nt) {
                const int col = waveN * 64 + nt * 16 + l16;
                const int kb  = kk * 32 + lhalf * 8;
                ABu u;
                u.h[0] = *(const v8bf*)&Bs[col * LDSTR + kb];
                u.h[1] = *(const v8bf*)&Bs[col * LDSTR + kb + 16];
                bfr[nt] = u.v;
            }
#pragma unroll
            for (int mt = 0; mt < 2; ++mt)
#pragma unroll
                for (int nt = 0; nt < 4; ++nt)
                    acc[mt][nt] = __builtin_amdgcn_wmma_f32_16x16x32_bf16(
                        false, afr[mt], false, bfr[nt],
                        (short)0, acc[mt][nt], false, false);
        }
        __syncthreads();
    }

    // ---- store: C/D layout -> VGPR v holds M = v + 8*lhalf, N = l16 ----
#pragma unroll
    for (int mt = 0; mt < 2; ++mt)
#pragma unroll
        for (int nt = 0; nt < 4; ++nt) {
            const long long row0 = rowBaseA + waveM * 32 + mt * 16 + 8 * lhalf;
            const long long col  = rowBaseB + waveN * 64 + nt * 16 + l16;
            float* cp = C + row0 * OUT_DIM + col;
#pragma unroll
            for (int v = 0; v < 8; ++v)
                cp[(long long)v * OUT_DIM] = acc[mt][nt][v];
        }
}

// ---------------------------------------------------------------------------
extern "C" void kernel_launch(void* const* d_in, const int* in_sizes, int n_in,
                              void* d_out, int out_size, void* d_ws, size_t ws_size,
                              hipStream_t stream) {
    (void)in_sizes; (void)n_in; (void)out_size; (void)ws_size;
    const float* x        = (const float*)d_in[0];
    const int*   qweight  = (const int*)d_in[1];
    const int*   qscalers = (const int*)d_in[2];
    const float* qfactor  = (const float*)d_in[3];
    const float* smean    = (const float*)d_in[4];
    float* out            = (float*)d_out;

    // workspace layout: [ W bf16 : 90,177,536 B ][ X bf16 : 67,108,864 B ]
    __bf16* wbf = (__bf16*)d_ws;
    __bf16* xbf = (__bf16*)((char*)d_ws + (size_t)OUT_DIM * IN_DIM * 2);

    // Pass 1: dequant weights (OUT*IN / 4 threads)
    nf4_dequant_kernel<<<(OUT_DIM * (long long)IN_DIM / 4 + 255) / 256, 256, 0, stream>>>(
        qweight, qscalers, qfactor, smean, wbf);
    // Pass 2: convert x (M*K / 4 threads)
    cvt_x_kernel<<<((long long)M_DIM * IN_DIM / 4 + 255) / 256, 256, 0, stream>>>(x, xbf);
    // Pass 3: GEMM, tiles: N 11008/128 = 86, M 8192/128 = 64
    nf4_gemm_wmma_kernel<<<dim3(OUT_DIM / 128, M_DIM / 128), 256, 0, stream>>>(
        xbf, wbf, out);
}
